// QuotientGraphVAE_84877143704151
// MI455X (gfx1250) — matrix operand order repaired and checked by reading, not compiled, not run on hardware
//
#include <hip/hip_runtime.h>
#include <hip/hip_bf16.h>
#include <math.h>

// ---------------------------------------------------------------------------
// CDNA5 (gfx1250) QuotientGraphVAE forward.
// Heavy GEMMs: v_wmma_f32_16x16x32_bf16, 2x2 tiles/wave, A staged via
// GLOBAL_LOAD_ASYNC_TO_LDS_B128 (ASYNCcnt) then converted LDS->LDS to bf16.
// Weights converted+transposed to bf16 col-major once per launch (L2-resident).
// Graph softmax/aggregation: wave-per-edge atomics (wave32 shuffles).
// ---------------------------------------------------------------------------

typedef __attribute__((ext_vector_type(16))) __bf16       v16bf;
typedef __attribute__((ext_vector_type(8)))  float        v8f;
typedef __attribute__((ext_vector_type(8)))  unsigned int v8u;
typedef __attribute__((address_space(3)))    char         lds_char;

#define HID 256
#define LAT 128
#define MAXN 100

__device__ __forceinline__ unsigned short f2bf(float f) {
  unsigned int x = __float_as_uint(f);
  unsigned int r = x + 0x7fffu + ((x >> 16) & 1u);   // round-to-nearest-even
  return (unsigned short)(r >> 16);
}

__device__ __forceinline__ float siluf(float v) { return v / (1.0f + __expf(-v)); }

__device__ __forceinline__ void atomicMaxF(float* addr, float val) {
  unsigned int* u = (unsigned int*)addr;
  unsigned int old = *u;
  while (__uint_as_float(old) < val) {
    unsigned int assumed = old;
    old = atomicCAS(u, assumed, __float_as_uint(val));
    if (old == assumed) break;
  }
}

// ---------------------------------------------------------------------------
// Generic fills
// ---------------------------------------------------------------------------
__global__ void k_fill(float* p, long cnt, float v) {
  long i = (long)blockIdx.x * blockDim.x + threadIdx.x;
  if (i < cnt) p[i] = v;
}

// ---------------------------------------------------------------------------
// Weight convert: W[K x N] fp32 row-major -> Wt[N x K] bf16 (col-major of W)
// ---------------------------------------------------------------------------
__global__ void k_convert_w(const float* __restrict__ W, unsigned short* __restrict__ Wt,
                            int K, int N) {
  long i = (long)blockIdx.x * blockDim.x + threadIdx.x;
  long tot = (long)K * N;
  if (i >= tot) return;
  int n = (int)(i / K), k = (int)(i - (long)n * K);
  Wt[i] = f2bf(W[(long)k * N + n]);
}

// ---------------------------------------------------------------------------
// WMMA GEMM: C[M x N] = act(A[M x K] * W[K x N] + bias).
// block = 128 threads (4 waves) -> 32-row x 128-col slab; each wave owns a
// 2x2 grid of 16x16 tiles (4 accumulators), so each A/B fragment feeds 2 WMMAs.
// A slab (fp32, contiguous 32*K floats) pulled into LDS with
// global_load_async_to_lds_b128, s_wait_asynccnt, then converted to bf16 once.
// K multiple of 32, K<=256; M multiple of 32.  grid = (M/32, ceil(N/128)).
// ---------------------------------------------------------------------------
__global__ void k_gemm_wmma(const float* __restrict__ A, const unsigned short* __restrict__ Wt,
                            const float* __restrict__ bias, float* __restrict__ C,
                            int M, int K, int N, int ldc, int act) {
  __shared__ __align__(16) float          fA[32 * 256];
  __shared__ __align__(16) unsigned short lA[32 * 256];
  const int  tid = threadIdx.x;
  const long m0  = (long)blockIdx.x * 32;

  // --- async DMA: 32 rows x K fp32 (contiguous slab) -> LDS ---
  {
    const float* gbase = A + m0 * (long)K;
    const int nchunks = (32 * K) / 4;                  // float4 chunks
    for (int c = tid; c < nchunks; c += 128) {
      unsigned la = (unsigned)(unsigned long long)(lds_char*)(&fA[c * 4]);
      unsigned long long ga = (unsigned long long)(gbase + c * 4);
      asm volatile("global_load_async_to_lds_b128 %0, %1, off"
                   :: "v"(la), "v"(ga) : "memory");
    }
    asm volatile("s_wait_asynccnt 0x0" ::: "memory");
  }
  __syncthreads();
  // --- one-shot fp32 -> bf16 conversion (amortized over K-loop x 8 tiles) ---
  for (int i = tid; i < 32 * K; i += 128) lA[i] = f2bf(fA[i]);
  __syncthreads();

  const int wave = tid >> 5;
  const int lane = tid & 31;
  const int n0 = blockIdx.y * 128 + wave * 32;
  if (n0 >= N) return;

  const int hi   = lane >> 4;                // lane-group: 0 = lanes 0-15, 1 = 16-31
  const int col0 = n0 + (lane & 15);
  const int col1 = col0 + 16;
  const float b0v = bias ? bias[col0] : 0.0f;
  const float b1v = bias ? bias[col1] : 0.0f;
  v8f acc00, acc01, acc10, acc11;
#pragma unroll
  for (int i = 0; i < 8; i++) { acc00[i] = b0v; acc01[i] = b1v; acc10[i] = b0v; acc11[i] = b1v; }

  const unsigned short* w0 = Wt + (long)col0 * K;
  const unsigned short* w1 = Wt + (long)col1 * K;
  __builtin_prefetch(w0, 0, 0);
  __builtin_prefetch(w1, 0, 0);
  const int ar0 = (lane & 15) * K;           // row tile 0 (rows 0..15)
  const int ar1 = ar0 + 16 * K;              // row tile 1 (rows 16..31)

  for (int ks = 0; ks < K; ks += 32) {
    // A fragments (CDNA5 16-bit A layout): lane<16 -> K {ks..+7, ks+16..+23}; lane>=16 -> +8
    const int ab0 = ar0 + ks + (hi << 3);
    const int ab1 = ar1 + ks + (hi << 3);
    uint4 a00 = *(const uint4*)(&lA[ab0]);
    uint4 a01 = *(const uint4*)(&lA[ab0 + 16]);
    uint4 a10 = *(const uint4*)(&lA[ab1]);
    uint4 a11 = *(const uint4*)(&lA[ab1 + 16]);
    v8u arx; arx[0]=a00.x; arx[1]=a00.y; arx[2]=a00.z; arx[3]=a00.w;
             arx[4]=a01.x; arx[5]=a01.y; arx[6]=a01.z; arx[7]=a01.w;
    v8u ary; ary[0]=a10.x; ary[1]=a10.y; ary[2]=a10.z; ary[3]=a10.w;
             ary[4]=a11.x; ary[5]=a11.y; ary[6]=a11.z; ary[7]=a11.w;
    // B fragments (same layout as A): lane<16 -> K {ks..+7, ks+16..+23}; lane>=16 -> +8
    const uint4* bp0 = (const uint4*)(w0 + ks + (hi << 3));
    const uint4* bp1 = (const uint4*)(w1 + ks + (hi << 3));
    uint4 b00 = bp0[0], b01 = bp0[2];
    uint4 b10 = bp1[0], b11 = bp1[2];
    v8u brx; brx[0]=b00.x; brx[1]=b00.y; brx[2]=b00.z; brx[3]=b00.w;
             brx[4]=b01.x; brx[5]=b01.y; brx[6]=b01.z; brx[7]=b01.w;
    v8u bry; bry[0]=b10.x; bry[1]=b10.y; bry[2]=b10.z; bry[3]=b10.w;
             bry[4]=b11.x; bry[5]=b11.y; bry[6]=b11.z; bry[7]=b11.w;
    v16bf af0 = __builtin_bit_cast(v16bf, arx);
    v16bf af1 = __builtin_bit_cast(v16bf, ary);
    v16bf bf0 = __builtin_bit_cast(v16bf, brx);
    v16bf bf1 = __builtin_bit_cast(v16bf, bry);
    acc00 = __builtin_amdgcn_wmma_f32_16x16x32_bf16(false, af0, false, bf0, (short)0, acc00, false, false);
    acc01 = __builtin_amdgcn_wmma_f32_16x16x32_bf16(false, af0, false, bf1, (short)0, acc01, false, false);
    acc10 = __builtin_amdgcn_wmma_f32_16x16x32_bf16(false, af1, false, bf0, (short)0, acc10, false, false);
    acc11 = __builtin_amdgcn_wmma_f32_16x16x32_bf16(false, af1, false, bf1, (short)0, acc11, false, false);
  }

#pragma unroll
  for (int r = 0; r < 8; r++) {
    long row0 = m0 + r + (hi << 3);
    long row1 = row0 + 16;
    float v00 = acc00[r], v01 = acc01[r], v10 = acc10[r], v11 = acc11[r];
    if (act == 1) { v00=siluf(v00); v01=siluf(v01); v10=siluf(v10); v11=siluf(v11); }
    else if (act == 2) { v00=fmaxf(v00,0.f); v01=fmaxf(v01,0.f); v10=fmaxf(v10,0.f); v11=fmaxf(v11,0.f); }
    C[row0 * (long)ldc + col0] = v00;
    C[row0 * (long)ldc + col1] = v01;
    C[row1 * (long)ldc + col0] = v10;
    C[row1 * (long)ldc + col1] = v11;
  }
}

// ---------------------------------------------------------------------------
// Node encoder pieces
// ---------------------------------------------------------------------------
__global__ void k_node_embed(const float* __restrict__ nf, const float* __restrict__ emb,
                             float* __restrict__ xcat, int n) {
  __shared__ int sidx;
  int i = blockIdx.x;
  if (i >= n) return;
  if (threadIdx.x == 0) {
    const float* r = nf + (long)i * 103;
    int best = 0; float bv = r[0];
    for (int k = 1; k < 100; k++) { float v = r[k]; if (v > bv) { bv = v; best = k; } }
    sidx = best;
  }
  __syncthreads();
  if (threadIdx.x < 64) xcat[(long)i * 128 + threadIdx.x] = emb[(long)sidx * 64 + threadIdx.x];
}

__global__ void k_coord1(const float* __restrict__ nf, const float* __restrict__ W,
                         const float* __restrict__ B, float* __restrict__ h1, long n) {
  long idx = (long)blockIdx.x * blockDim.x + threadIdx.x;
  if (idx >= n * 256) return;
  long i = idx >> 8; int j = (int)(idx & 255);
  const float* c = nf + i * 103 + 100;
  float s = B[j] + c[0] * W[j] + c[1] * W[256 + j] + c[2] * W[512 + j];
  h1[idx] = siluf(s);
}

// ---------------------------------------------------------------------------
// LayerNorm (+ optional silu), wave per row, in place.  d multiple of 32.
// ---------------------------------------------------------------------------
__global__ void k_ln_silu(float* __restrict__ X, const float* __restrict__ g,
                          const float* __restrict__ b, long M, int d, int dosilu) {
  int wave = threadIdx.x >> 5, lane = threadIdx.x & 31;
  long row = (long)blockIdx.x * 8 + wave;
  if (row >= M) return;
  float* x = X + row * (long)d;
  float s = 0.0f, sq = 0.0f;
  for (int j = lane; j < d; j += 32) { float v = x[j]; s += v; sq += v * v; }
  for (int o = 16; o >= 1; o >>= 1) { s += __shfl_xor(s, o, 32); sq += __shfl_xor(sq, o, 32); }
  float mu = s / d;
  float var = sq / d - mu * mu;
  float inv = rsqrtf(var + 1e-5f);
  for (int j = lane; j < d; j += 32) {
    float v = (x[j] - mu) * inv * g[j] + b[j];
    if (dosilu) v = siluf(v);
    x[j] = v;
  }
}

// ---------------------------------------------------------------------------
// Fused edge MLP: dist1/per1 (tiny-K, silu) + dist2/per2 -> ec_in[E x 64]
// ---------------------------------------------------------------------------
__global__ void k_edge_enc(const float* __restrict__ ef,
                           const float* __restrict__ d1w, const float* __restrict__ d1b,
                           const float* __restrict__ p1w, const float* __restrict__ p1b,
                           const float* __restrict__ d2w, const float* __restrict__ d2b,
                           const float* __restrict__ p2w, const float* __restrict__ p2b,
                           float* __restrict__ ecin, long E) {
  __shared__ float hd[128], hp[128];
  long e = blockIdx.x;
  if (e >= E) return;
  int t = threadIdx.x;
  float e0 = ef[e * 4 + 0];
  float q0 = ef[e * 4 + 1], q1 = ef[e * 4 + 2], q2 = ef[e * 4 + 3];
  hd[t] = siluf(e0 * d1w[t] + d1b[t]);
  hp[t] = siluf(q0 * p1w[t] + q1 * p1w[128 + t] + q2 * p1w[256 + t] + p1b[t]);
  __syncthreads();
  if (t < 64) {
    int j = t & 31;
    const bool isP = t >= 32;
    const float* W = isP ? p2w : d2w;
    const float* B = isP ? p2b : d2b;
    const float* h = isP ? hp : hd;
    float s = B[j];
#pragma unroll 4
    for (int k = 0; k < 128; k++) s += h[k] * W[k * 32 + j];
    ecin[e * 64 + t] = s;     // de in cols 0..31, pe in 32..63
  }
}

// ---------------------------------------------------------------------------
// Self-loop attr = mean of incoming edge attrs (PyG fill_value='mean')
// ---------------------------------------------------------------------------
__global__ void k_loop_acc(const int* __restrict__ ei, float* __restrict__ e2,
                           float* __restrict__ deg, long E) {
  long e = blockIdx.x;
  if (e >= E) return;
  int d = ei[E + e];
  if (threadIdx.x == 0) atomicAdd(&deg[d], 1.0f);
  atomicAdd(&e2[((long)E + d) * 256 + threadIdx.x], e2[e * 256 + threadIdx.x]);
}

__global__ void k_loop_div(float* __restrict__ e2, const float* __restrict__ deg,
                           long E, long n) {
  long idx = (long)blockIdx.x * blockDim.x + threadIdx.x;
  if (idx >= n * 256) return;
  long i = idx >> 8;
  e2[(E + i) * 256 + (idx & 255)] /= fmaxf(deg[i], 1.0f);
}

// ---------------------------------------------------------------------------
// GATv2: logits (segment max) -> exp (segment sum) -> aggregate -> finalize
// ---------------------------------------------------------------------------
__global__ void k_gat_logits(const float* __restrict__ XL, const float* __restrict__ XR,
                             const float* __restrict__ EL, const int* __restrict__ ei,
                             const float* __restrict__ att, float* __restrict__ a,
                             float* __restrict__ amax, long E, long E2, int heads) {
  int wave = threadIdx.x >> 5, lane = threadIdx.x & 31;
  long e = (long)blockIdx.x * 8 + wave;
  if (e >= E2) return;
  int s, d;
  if (e < E) { s = ei[e]; d = ei[E + e]; } else { s = d = (int)(e - E); }
  const float* xl = XL + (long)s * 256;
  const float* xr = XR + (long)d * 256;
  const float* el = EL + e * 256;
  float part = 0.0f;
  const int base = lane * 8;
#pragma unroll
  for (int t = 0; t < 8; t++) {
    int hc = base + t;
    float m = xl[hc] + xr[hc] + el[hc];
    m = (m > 0.0f) ? m : 0.2f * m;          // leaky_relu(0.2)
    part += m * att[hc];
  }
  const int lph = 32 / heads;               // lanes per head (8 or 16)
  for (int o = 1; o < lph; o <<= 1) part += __shfl_xor(part, o, 32);
  if ((lane & (lph - 1)) == 0) {
    int h = lane / lph;
    a[e * heads + h] = part;
    atomicMaxF(&amax[(long)d * heads + h], part);
  }
}

__global__ void k_gat_exp(float* __restrict__ a, const float* __restrict__ amax,
                          float* __restrict__ asum, const int* __restrict__ ei,
                          long E, long E2, int heads) {
  long idx = (long)blockIdx.x * blockDim.x + threadIdx.x;
  if (idx >= E2 * heads) return;
  long e = idx / heads; int h = (int)(idx - e * heads);
  int d = (e < E) ? ei[E + e] : (int)(e - E);
  float p = __expf(a[idx] - amax[(long)d * heads + h]);
  a[idx] = p;
  atomicAdd(&asum[(long)d * heads + h], p);
}

__global__ void k_gat_aggr(const float* __restrict__ XL, const float* __restrict__ a,
                           const float* __restrict__ asum, const int* __restrict__ ei,
                           float* __restrict__ agg, long E, long E2, int heads, int ch) {
  int wave = threadIdx.x >> 5, lane = threadIdx.x & 31;
  long e = (long)blockIdx.x * 8 + wave;
  if (e >= E2) return;
  int s, d;
  if (e < E) { s = ei[e]; d = ei[E + e]; } else { s = d = (int)(e - E); }
  const int base = lane * 8;
  const int h = base / ch;
  const float alpha = a[e * heads + h] / (asum[(long)d * heads + h] + 1e-16f);
  const float* xl = XL + (long)s * 256;
  float* out = agg + (long)d * 256;
#pragma unroll
  for (int t = 0; t < 8; t++) {
    int hc = base + t;
    atomicAdd(&out[hc], xl[hc] * alpha);
  }
}

__global__ void k_gat_final(const float* __restrict__ agg, const float* __restrict__ bias,
                            float* __restrict__ x, long n) {
  long idx = (long)blockIdx.x * blockDim.x + threadIdx.x;
  if (idx >= n * 256) return;
  x[idx] = fmaxf(agg[idx] + bias[idx & 255], 0.0f);
}

// ---------------------------------------------------------------------------
// Column mean pooling: out[c] += scale * sum_rows X[r,c]   (256 cols)
// ---------------------------------------------------------------------------
__global__ void k_colsum(const float* __restrict__ X, float* __restrict__ out,
                         long M, float scale) {
  int t = threadIdx.x;
  float s = 0.0f;
  for (long r = blockIdx.x; r < M; r += gridDim.x) s += X[r * 256 + t];
  atomicAdd(&out[t], s * scale);
}

// ---------------------------------------------------------------------------
// Batch-1 dense: out[j] = act(b[j] + sum_k in[k]*W[k*N+j])
// ---------------------------------------------------------------------------
__global__ void k_dense(const float* __restrict__ in, const float* __restrict__ W,
                        const float* __restrict__ b, float* __restrict__ out,
                        int K, int N, int act) {
  int j = blockIdx.x * blockDim.x + threadIdx.x;
  if (j >= N) return;
  float s = b ? b[j] : 0.0f;
  for (int k = 0; k < K; k++) s += in[k] * W[(long)k * N + j];
  if (act == 1) s = siluf(s);
  out[j] = s;
}

__global__ void k_z(const float* __restrict__ mu, const float* __restrict__ lv,
                    const float* __restrict__ eps, float* __restrict__ z,
                    float* __restrict__ out) {
  int j = threadIdx.x;
  if (j >= LAT) return;
  float m = mu[j], l = lv[j];
  z[j] = m + eps[j] * __expf(0.5f * l);
  out[20918 + j] = m;   // mu output
  out[21046 + j] = l;   // lv output
}

__global__ void k_copy(const float* __restrict__ src, float* __restrict__ dst, int n) {
  int j = blockIdx.x * blockDim.x + threadIdx.x;
  if (j < n) dst[j] = src[j];
}

__global__ void k_pack_nf(const float* __restrict__ atom, const float* __restrict__ crd,
                          float* __restrict__ out) {
  int i = blockIdx.x * blockDim.x + threadIdx.x;
  if (i >= MAXN * 103) return;
  int r = i / 103, c = i - r * 103;
  out[i] = (c < 100) ? atom[r * 100 + c] : crd[r * 3 + (c - 100)];
}

__global__ void k_cell(const float* __restrict__ cp, float* __restrict__ out) {
  int j = threadIdx.x;
  if (j >= 6) return;
  float x = cp[j];
  float v;
  if (j < 3) v = (x > 20.0f) ? x : log1pf(__expf(x));          // softplus
  else       v = 30.0f + 120.0f / (1.0f + __expf(-x));          // 30 + 120*sigmoid
  out[20400 + j] = v;
}

// ---------------------------------------------------------------------------
// Host launch
// ---------------------------------------------------------------------------
extern "C" void kernel_launch(void* const* d_in, const int* in_sizes, int n_in,
                              void* d_out, int out_size, void* d_ws, size_t ws_size,
                              hipStream_t stream) {
  if (n_in < 82 || !d_ws) return;
  const long n  = in_sizes[0] / 103;     // 20000
  const long E  = in_sizes[1] / 2;       // 320000
  const long E2 = E + n;                 // 340000

  const float* node_features = (const float*)d_in[0];
  const int*   edge_index    = (const int*)d_in[1];     // [0..E)=src, [E..2E)=dst
  const float* edge_features = (const float*)d_in[2];
  const float* eps           = (const float*)d_in[3];

  // params: jax pytree leaves in sorted-dict-key order (None biases dropped)
  int pi = 4;
  auto P = [&](void) -> const float* { return (const float*)d_in[pi++]; };
  const float *coord1_b=P(), *coord1_w=P(), *coord2_b=P(), *coord2_w=P();
  const float *atom1_b=P(), *atom1_w=P(), *atom2_b=P(), *atom2_w=P();
  const float *cell1_b=P(), *cell1_w=P(), *cell2_b=P(), *cell2_w=P();
  const float *crd1_b=P(),  *crd1_w=P(),  *crd2_b=P(),  *crd2_w=P();
  const float *edge1_b=P(), *edge1_w=P(), *edge2_b=P(), *edge2_w=P();
  const float *l2h1_b=P(),  *l2h1_w=P(),  *l2h2_b=P(),  *l2h2_w=P();
  const float *ln1_b=P(), *ln1_g=P(), *ln2_b=P(), *ln2_g=P();
  const float *nn1_b=P(), *nn1_w=P(), *nn2_b=P(), *nn2_w=P();
  const float *dist1_b=P(), *dist1_w=P(), *dist2_b=P(), *dist2_w=P();
  const float *ecomb_b=P(), *ecomb_w=P(), *eln_b=P(), *eln_g=P();
  const float *emb=P();
  const float *final_b=P(), *final_w=P(), *fln_b=P(), *fln_g=P();
  const float *g_att[3], *g_bias[3], *g_we[3], *g_bl[3], *g_wl[3], *g_br[3], *g_wr[3];
  for (int L = 0; L < 3; L++) {
    g_att[L]=P(); g_bias[L]=P(); g_we[L]=P(); g_bl[L]=P(); g_wl[L]=P(); g_br[L]=P(); g_wr[L]=P();
  }
  const float *lv_b=P(), *lv_w=P(), *mu_b=P(), *mu_w=P();
  const float *ncomb_b=P(), *ncomb_w=P(), *nln_b=P(), *nln_g=P();
  const float *per1_b=P(), *per1_w=P(), *per2_b=P(), *per2_w=P();

  // ---------------- workspace layout ----------------
  char* base = (char*)d_ws; size_t off = 0;
  auto ALLOC = [&](size_t bytes) -> void* {
    void* p = base + off; off = (off + bytes + 255) & ~(size_t)255; return p;
  };
  float* f_x    = (float*)ALLOC((size_t)n * 256 * 4);
  float* f_XL   = (float*)ALLOC((size_t)n * 256 * 4);
  float* f_XR   = (float*)ALLOC((size_t)n * 256 * 4);
  float* f_agg  = (float*)ALLOC((size_t)n * 256 * 4);   // also coord hidden h1
  float* f_xcat = (float*)ALLOC((size_t)n * 128 * 4);
  float* f_e2   = (float*)ALLOC((size_t)E2 * 256 * 4);
  float* f_EL   = (float*)ALLOC((size_t)E2 * 256 * 4);
  float* f_ecin = (float*)ALLOC((size_t)E * 64 * 4);
  float* f_a    = (float*)ALLOC((size_t)E2 * 4 * 4);
  float* f_amax = (float*)ALLOC((size_t)n * 4 * 4);
  float* f_asum = (float*)ALLOC((size_t)n * 4 * 4);
  float* f_deg  = (float*)ALLOC((size_t)n * 4);
  float* sv     = (float*)ALLOC(65536 * 4);             // batch-1 small vectors
  unsigned short* us_coord2 = (unsigned short*)ALLOC((size_t)64 * 256 * 2);
  unsigned short* us_ncomb  = (unsigned short*)ALLOC((size_t)256 * 128 * 2);
  unsigned short* us_ecomb  = (unsigned short*)ALLOC((size_t)256 * 64 * 2);
  unsigned short* us_gl[3], *us_gr[3], *us_ge[3];
  for (int L = 0; L < 3; L++) {
    us_gl[L] = (unsigned short*)ALLOC((size_t)256 * 256 * 2);
    us_gr[L] = (unsigned short*)ALLOC((size_t)256 * 256 * 2);
    us_ge[L] = (unsigned short*)ALLOC((size_t)256 * 256 * 2);
  }
  if (off > ws_size) return;   // insufficient scratch

  float* g_   = sv + 0;      // 512
  float* gf   = sv + 512;    // 256
  float* muv  = sv + 768;    // 128
  float* lvv  = sv + 896;    // 128
  float* zv   = sv + 1024;   // 128
  float* h1d  = sv + 1152;   // 256
  float* h2d  = sv + 1408;   // 512
  float* t256 = sv + 1920;   // 256
  float* t512 = sv + 2176;   // 512
  float* atomv= sv + 2688;   // 10000
  float* crdv = sv + 12688;  // 300
  float* cpv  = sv + 12992;  // 6

  float* out = (float*)d_out;
  const dim3 B256(256), B128(128);
  auto G = [](long work, int blk) { return dim3((unsigned)((work + blk - 1) / blk)); };

  // ---------------- weight conversion (bf16, transposed) ----------------
  k_convert_w<<<G(256L*64,256), B256, 0, stream>>>(coord2_w, us_coord2, 256, 64);
  k_convert_w<<<G(128L*256,256), B256, 0, stream>>>(ncomb_w,  us_ncomb, 128, 256);
  k_convert_w<<<G(64L*256,256),  B256, 0, stream>>>(ecomb_w,  us_ecomb,  64, 256);
  for (int L = 0; L < 3; L++) {
    k_convert_w<<<G(256L*256,256), B256, 0, stream>>>(g_wl[L], us_gl[L], 256, 256);
    k_convert_w<<<G(256L*256,256), B256, 0, stream>>>(g_wr[L], us_gr[L], 256, 256);
    k_convert_w<<<G(256L*256,256), B256, 0, stream>>>(g_we[L], us_ge[L], 256, 256);
  }

  // ---------------- node encoder ----------------
  k_node_embed<<<dim3((unsigned)n), dim3(64), 0, stream>>>(node_features, emb, f_xcat, (int)n);
  k_coord1<<<G(n*256,256), B256, 0, stream>>>(node_features, coord1_w, coord1_b, f_agg, n);
  // cf = coord2(h1) -> xcat[:,64:128]
  k_gemm_wmma<<<dim3((unsigned)(n/32), 1), B128, 0, stream>>>(
      f_agg, us_coord2, coord2_b, f_xcat + 64, (int)n, 256, 64, 128, 0);
  // x = silu(LN(ncomb(xcat)))
  k_gemm_wmma<<<dim3((unsigned)(n/32), 2), B128, 0, stream>>>(
      f_xcat, us_ncomb, ncomb_b, f_x, (int)n, 128, 256, 256, 0);
  k_ln_silu<<<G(n,8), B256, 0, stream>>>(f_x, nln_g, nln_b, n, 256, 1);

  // ---------------- edge encoder ----------------
  k_edge_enc<<<dim3((unsigned)E), B128, 0, stream>>>(
      edge_features, dist1_w, dist1_b, per1_w, per1_b,
      dist2_w, dist2_b, per2_w, per2_b, f_ecin, E);
  k_gemm_wmma<<<dim3((unsigned)(E/32), 2), B128, 0, stream>>>(
      f_ecin, us_ecomb, ecomb_b, f_e2, (int)E, 64, 256, 256, 0);
  k_ln_silu<<<G(E,8), B256, 0, stream>>>(f_e2, eln_g, eln_b, E, 256, 1);

  // self loops: loop attr = mean of incoming eattr
  k_fill<<<G(n*256,256), B256, 0, stream>>>(f_e2 + (size_t)E * 256, n * 256, 0.0f);
  k_fill<<<G(n,256), B256, 0, stream>>>(f_deg, n, 0.0f);
  k_loop_acc<<<dim3((unsigned)E), B256, 0, stream>>>(edge_index, f_e2, f_deg, E);
  k_loop_div<<<G(n*256,256), B256, 0, stream>>>(f_e2, f_deg, E, n);

  // ---------------- 3 x GATv2 ----------------
  const int heads_arr[3] = {4, 4, 2};
  for (int L = 0; L < 3; L++) {
    const int heads = heads_arr[L], ch = 256 / heads;
    k_gemm_wmma<<<dim3((unsigned)(n/32), 2), B128, 0, stream>>>(
        f_x, us_gl[L], g_bl[L], f_XL, (int)n, 256, 256, 256, 0);
    k_gemm_wmma<<<dim3((unsigned)(n/32), 2), B128, 0, stream>>>(
        f_x, us_gr[L], g_br[L], f_XR, (int)n, 256, 256, 256, 0);
    k_gemm_wmma<<<dim3((unsigned)(E2/32), 2), B128, 0, stream>>>(
        f_e2, us_ge[L], nullptr, f_EL, (int)E2, 256, 256, 256, 0);
    k_fill<<<G(n*heads,256), B256, 0, stream>>>(f_amax, n * heads, -1e30f);
    k_fill<<<G(n*heads,256), B256, 0, stream>>>(f_asum, n * heads, 0.0f);
    k_fill<<<G(n*256,256), B256, 0, stream>>>(f_agg, n * 256, 0.0f);
    k_gat_logits<<<G(E2,8), B256, 0, stream>>>(f_XL, f_XR, f_EL, edge_index,
                                               g_att[L], f_a, f_amax, E, E2, heads);
    k_gat_exp<<<G(E2*heads,256), B256, 0, stream>>>(f_a, f_amax, f_asum, edge_index,
                                                    E, E2, heads);
    k_gat_aggr<<<G(E2,8), B256, 0, stream>>>(f_XL, f_a, f_asum, edge_index,
                                             f_agg, E, E2, heads, ch);
    k_gat_final<<<G(n*256,256), B256, 0, stream>>>(f_agg, g_bias[L], f_x, n);
  }

  // ---------------- pooling + latent ----------------
  k_fill<<<G(512,256), B256, 0, stream>>>(g_, 512, 0.0f);
  k_colsum<<<dim3(256), B256, 0, stream>>>(f_x, g_, n, 1.0f / (float)n);
  k_colsum<<<dim3(256), B256, 0, stream>>>(f_e2, g_ + 256, E, 1.0f / (float)E);
  k_dense<<<G(256,256), B256, 0, stream>>>(g_, final_w, final_b, gf, 512, 256, 0);
  k_ln_silu<<<dim3(1), B256, 0, stream>>>(gf, fln_g, fln_b, 1, 256, 1);
  k_dense<<<G(128,128), B128, 0, stream>>>(gf, mu_w, mu_b, muv, 256, 128, 0);
  k_dense<<<G(128,128), B128, 0, stream>>>(gf, lv_w, lv_b, lvv, 256, 128, 0);
  k_z<<<dim3(1), B128, 0, stream>>>(muv, lvv, eps, zv, out);

  // ---------------- decoder ----------------
  k_dense<<<G(256,256), B256, 0, stream>>>(zv, l2h1_w, l2h1_b, h1d, 128, 256, 0);
  k_ln_silu<<<dim3(1), B256, 0, stream>>>(h1d, ln1_g, ln1_b, 1, 256, 1);
  k_dense<<<G(512,256), B256, 0, stream>>>(h1d, l2h2_w, l2h2_b, h2d, 256, 512, 0);
  k_ln_silu<<<dim3(1), B256, 0, stream>>>(h2d, ln2_g, ln2_b, 1, 512, 1);
  k_copy<<<G(512,256), B256, 0, stream>>>(h2d, out + 20406, 512);          // h

  k_dense<<<G(256,256), B256, 0, stream>>>(h2d, atom1_w, atom1_b, t256, 512, 256, 1);
  k_dense<<<G(10000,256), B256, 0, stream>>>(t256, atom2_w, atom2_b, atomv, 256, 10000, 0);
  k_dense<<<G(256,256), B256, 0, stream>>>(h2d, crd1_w, crd1_b, t256, 512, 256, 1);
  k_dense<<<G(300,256), B256, 0, stream>>>(t256, crd2_w, crd2_b, crdv, 256, 300, 0);
  k_pack_nf<<<G(10300,256), B256, 0, stream>>>(atomv, crdv, out);          // nf

  k_dense<<<G(512,256), B256, 0, stream>>>(h2d, edge1_w, edge1_b, t512, 512, 512, 1);
  k_dense<<<G(10000,256), B256, 0, stream>>>(t512, edge2_w, edge2_b, out + 10300,
                                             512, 10000, 0);               // el_
  k_dense<<<G(256,256), B256, 0, stream>>>(h2d, nn1_w, nn1_b, t256, 512, 256, 1);
  k_dense<<<G(100,128), B128, 0, stream>>>(t256, nn2_w, nn2_b, out + 20300, 256, 100, 0); // nnl
  k_dense<<<G(256,256), B256, 0, stream>>>(h2d, cell1_w, cell1_b, t256, 512, 256, 1);
  k_dense<<<G(6,32), dim3(32), 0, stream>>>(t256, cell2_w, cell2_b, cpv, 256, 6, 0);
  k_cell<<<dim3(1), dim3(32), 0, stream>>>(cpv, out);                      // cell
}
